// PRMPConv_1099511628113
// MI455X (gfx1250) — matrix operand order repaired and checked by reading, not compiled, not run on hardware
//
#include <hip/hip_runtime.h>

typedef __attribute__((ext_vector_type(16))) __bf16 v16bf;
typedef __attribute__((ext_vector_type(8)))  float  v8f;

#define HID     128
#define NPARENT 100000
#define NCHILD  200000
#define NEDGES  600000
#define LN_EPS  1e-5f

// ---------------------------------------------------------------------------
// A-fragment (16x32 bf16) gather from a 128-float row in global memory.
// Lane layout per ISA 7.12.2: lanes 0-15 hold row M=lane, K = {0..7,16..23}+kt*32
//                             lanes 16-31 hold row M=lane-16, K = {8..15,24..31}+kt*32
// ---------------------------------------------------------------------------
__device__ __forceinline__ v16bf fragA_from_row(const float* __restrict__ row,
                                                int kt, int hsel, float scale) {
  const int k0 = kt * 32 + hsel * 8;
  float4 a = *(const float4*)(row + k0);
  float4 b = *(const float4*)(row + k0 + 4);
  float4 c = *(const float4*)(row + k0 + 16);
  float4 d = *(const float4*)(row + k0 + 20);
  v16bf f;
  f[0]  = (__bf16)(a.x * scale); f[1]  = (__bf16)(a.y * scale);
  f[2]  = (__bf16)(a.z * scale); f[3]  = (__bf16)(a.w * scale);
  f[4]  = (__bf16)(b.x * scale); f[5]  = (__bf16)(b.y * scale);
  f[6]  = (__bf16)(b.z * scale); f[7]  = (__bf16)(b.w * scale);
  f[8]  = (__bf16)(c.x * scale); f[9]  = (__bf16)(c.y * scale);
  f[10] = (__bf16)(c.z * scale); f[11] = (__bf16)(c.w * scale);
  f[12] = (__bf16)(d.x * scale); f[13] = (__bf16)(d.y * scale);
  f[14] = (__bf16)(d.z * scale); f[15] = (__bf16)(d.w * scale);
  return f;
}

// ---------------------------------------------------------------------------
// Pack a 128x128 f32 weight (row-major W[k][n]) into LDS as bf16 B-fragments.
// Fragment (kt,nt) covers K=kt*32..+31, N=nt*16..+15.
// B layout: lane n' (0-15) holds column n', K = kt*32 + {0..7,16..23} (elements 0..15);
//           lane n'+16 holds column n', K = kt*32 + {8..15,24..31}.
// LDS layout: frag-major, ((kt*8+nt)*32 + lane)*16 + i  (32B-aligned per lane)
// ---------------------------------------------------------------------------
__device__ __forceinline__ void pack_weight_lds(const float* __restrict__ W,
                                                __bf16* dstLds, int tid, int nthr) {
  for (int e = tid; e < 16384; e += nthr) {
    int frag = e >> 9;
    int r    = e & 511;
    int ln   = r >> 4;
    int i    = r & 15;
    int kt   = frag >> 3;
    int nt   = frag & 7;
    int k    = kt * 32 + ((i >> 3) << 4) + ((ln >> 4) << 3) + (i & 7);
    int n    = nt * 16 + (ln & 15);
    dstLds[e] = (__bf16)W[k * HID + n];
  }
}

// ---------------------------------------------------------------------------
// Kernel 0: zero ssum + cnt workspace
// ---------------------------------------------------------------------------
__global__ void zero_ws_kernel(float* __restrict__ ws, long n) {
  long i = (long)blockIdx.x * blockDim.x + threadIdx.x;
  long stride = (long)gridDim.x * blockDim.x;
  for (; i < n; i += stride) ws[i] = 0.0f;
}

// ---------------------------------------------------------------------------
// Kernel 1: per-parent edge counts
// ---------------------------------------------------------------------------
__global__ void count_kernel(const int* __restrict__ src, float* __restrict__ cnt) {
  int i = blockIdx.x * blockDim.x + threadIdx.x;
  if (i < NEDGES)
    __hip_atomic_fetch_add(&cnt[src[i]], 1.0f, __ATOMIC_RELAXED, __HIP_MEMORY_SCOPE_AGENT);
}

#define WMMA_BF16(A, B, C) \
  __builtin_amdgcn_wmma_f32_16x16x32_bf16(false, (A), false, (B), (short)0, (C), false, false)

// ---------------------------------------------------------------------------
// Kernel 2: edge pipeline. One wave = one 16-edge tile.
//   h    = relu(xp[src] @ W1 + b1)        (16x128, WMMA bf16)
//   pred = h @ W2 + b2                    (16x128, WMMA bf16, h via LDS reshape)
//   res  = xc[dst] - pred ; atomic scatter-add into ssum[src]
// ---------------------------------------------------------------------------
__global__ void __launch_bounds__(256) edge_kernel(
    const float* __restrict__ xp, const float* __restrict__ xc,
    const int* __restrict__ src, const int* __restrict__ dst,
    const float* __restrict__ W1, const float* __restrict__ b1,
    const float* __restrict__ W2, const float* __restrict__ b2,
    float* __restrict__ ssum) {
  extern __shared__ char smemRaw[];
  __bf16* w1p = (__bf16*)smemRaw;        // 16384 bf16 = 32 KB
  __bf16* w2p = w1p + 16384;             // 32 KB
  const int tid  = threadIdx.x;
  const int lane = tid & 31;
  const int wave = tid >> 5;
  __bf16* hst = w2p + 16384 + wave * 2048;   // per-wave 16x128 bf16 stage (4 KB)

  pack_weight_lds(W1, w1p, tid, blockDim.x);
  pack_weight_lds(W2, w2p, tid, blockDim.x);
  __syncthreads();

  const int hsel = lane >> 4;
  const int lrow = lane & 15;
  const int nTiles  = NEDGES / 16;           // 37500
  const int wstride = gridDim.x * 8;

  for (int tile = blockIdx.x * 8 + wave; tile < nTiles; tile += wstride) {
    const int eBase = tile * 16;

    // ---- build A fragments: gather one parent row per (lrow) pair of lanes
    const int p = src[eBase + lrow];
    const float* prow = xp + (size_t)p * HID;
    v16bf A0 = fragA_from_row(prow, 0, hsel, 1.0f);
    v16bf A1 = fragA_from_row(prow, 1, hsel, 1.0f);
    v16bf A2 = fragA_from_row(prow, 2, hsel, 1.0f);
    v16bf A3 = fragA_from_row(prow, 3, hsel, 1.0f);

    // ---- GEMM1 + bias + relu, emit h into per-wave LDS in A-fragment layout
    #pragma unroll
    for (int nt = 0; nt < 8; ++nt) {
      v8f c = {};
      v16bf B;
      B = *(const v16bf*)(w1p + (( 0 + nt) * 32 + lane) * 16); c = WMMA_BF16(A0, B, c);
      B = *(const v16bf*)(w1p + (( 8 + nt) * 32 + lane) * 16); c = WMMA_BF16(A1, B, c);
      B = *(const v16bf*)(w1p + ((16 + nt) * 32 + lane) * 16); c = WMMA_BF16(A2, B, c);
      B = *(const v16bf*)(w1p + ((24 + nt) * 32 + lane) * 16); c = WMMA_BF16(A3, B, c);

      const int   n   = nt * 16 + lrow;        // C layout: N = lane%16 (+16*nt)
      const float bb  = b1[n];
      const int   kk  = n & 31;                // h feeds GEMM2 with K = n
      const int   ktH = n >> 5;
      const int   lnOff = ((kk >> 3) & 1) << 4;
      const int   i2  = (kk & 7) + ((kk >> 4) << 3);
      #pragma unroll
      for (int v = 0; v < 8; ++v) {
        float hv = fmaxf(c[v] + bb, 0.0f);     // row M = v + 8*(lane/16)
        int m = v + (hsel << 3);
        hst[(ktH * 32 + m + lnOff) * 16 + i2] = (__bf16)hv;
      }
    }

    // LDS ops are in-order within a wave; wait + compiler barrier before reload
    asm volatile("s_wait_dscnt 0x0" ::: "memory");

    v16bf H0 = *(const v16bf*)(hst + (0 * 32 + lane) * 16);
    v16bf H1 = *(const v16bf*)(hst + (1 * 32 + lane) * 16);
    v16bf H2 = *(const v16bf*)(hst + (2 * 32 + lane) * 16);
    v16bf H3 = *(const v16bf*)(hst + (3 * 32 + lane) * 16);

    // src/dst indices for the 8 C-layout rows this lane owns
    int sv[8], dv[8];
    #pragma unroll
    for (int v = 0; v < 8; ++v) {
      int ee = eBase + v + (hsel << 3);
      sv[v] = src[ee];
      dv[v] = dst[ee];
    }

    // ---- GEMM2 + residual + atomic scatter-mean numerator
    #pragma unroll
    for (int nt = 0; nt < 8; ++nt) {
      v8f c = {};
      v16bf B;
      B = *(const v16bf*)(w2p + (( 0 + nt) * 32 + lane) * 16); c = WMMA_BF16(H0, B, c);
      B = *(const v16bf*)(w2p + (( 8 + nt) * 32 + lane) * 16); c = WMMA_BF16(H1, B, c);
      B = *(const v16bf*)(w2p + ((16 + nt) * 32 + lane) * 16); c = WMMA_BF16(H2, B, c);
      B = *(const v16bf*)(w2p + ((24 + nt) * 32 + lane) * 16); c = WMMA_BF16(H3, B, c);

      const int   n  = nt * 16 + lrow;
      const float bb = b2[n];
      #pragma unroll
      for (int v = 0; v < 8; ++v) {
        float pred = c[v] + bb;
        float res  = xc[(size_t)dv[v] * HID + n] - pred;
        __hip_atomic_fetch_add(&ssum[(size_t)sv[v] * HID + n], res,
                               __ATOMIC_RELAXED, __HIP_MEMORY_SCOPE_AGENT);
      }
    }
  }
}

// ---------------------------------------------------------------------------
// Kernel 3: parent update. agg = ssum/max(cnt,1); y = xp + agg@Wu + bu; LayerNorm.
// LayerNorm entirely in registers via width-16 shfl_xor butterflies (each
// 16-lane half-wave owns a disjoint set of 8 rows in the WMMA C layout).
// ---------------------------------------------------------------------------
__global__ void __launch_bounds__(256) parent_kernel(
    const float* __restrict__ xp, const float* __restrict__ ssum,
    const float* __restrict__ cnt,
    const float* __restrict__ Wu, const float* __restrict__ bu,
    const float* __restrict__ gamma, const float* __restrict__ beta,
    float* __restrict__ out) {
  extern __shared__ char smemRaw[];
  __bf16* wup = (__bf16*)smemRaw;        // 32 KB
  const int tid  = threadIdx.x;
  const int lane = tid & 31;
  const int wave = tid >> 5;

  pack_weight_lds(Wu, wup, tid, blockDim.x);
  __syncthreads();

  const int hsel = lane >> 4;
  const int lrow = lane & 15;
  const int nTiles  = NPARENT / 16;      // 6250
  const int wstride = gridDim.x * 8;

  for (int tile = blockIdx.x * 8 + wave; tile < nTiles; tile += wstride) {
    const int pBase = tile * 16;
    const int pr    = pBase + lrow;
    const float inv = 1.0f / fmaxf(cnt[pr], 1.0f);
    const float* row = ssum + (size_t)pr * HID;
    v16bf A0 = fragA_from_row(row, 0, hsel, inv);
    v16bf A1 = fragA_from_row(row, 1, hsel, inv);
    v16bf A2 = fragA_from_row(row, 2, hsel, inv);
    v16bf A3 = fragA_from_row(row, 3, hsel, inv);

    v8f acc[8];
    #pragma unroll
    for (int nt = 0; nt < 8; ++nt) {
      v8f c = {};
      v16bf B;
      B = *(const v16bf*)(wup + (( 0 + nt) * 32 + lane) * 16); c = WMMA_BF16(A0, B, c);
      B = *(const v16bf*)(wup + (( 8 + nt) * 32 + lane) * 16); c = WMMA_BF16(A1, B, c);
      B = *(const v16bf*)(wup + ((16 + nt) * 32 + lane) * 16); c = WMMA_BF16(A2, B, c);
      B = *(const v16bf*)(wup + ((24 + nt) * 32 + lane) * 16); c = WMMA_BF16(A3, B, c);
      acc[nt] = c;
    }

    #pragma unroll
    for (int v = 0; v < 8; ++v) {
      const int pm = pBase + v + (hsel << 3);
      const float* xrow = xp + (size_t)pm * HID;
      float yv[8];
      float sm = 0.0f, sq = 0.0f;
      #pragma unroll
      for (int nt = 0; nt < 8; ++nt) {
        const int n = nt * 16 + lrow;
        float y = xrow[n] + acc[nt][v] + bu[n];
        yv[nt] = y;
        sm += y;
        sq += y * y;
      }
      #pragma unroll
      for (int m = 1; m < 16; m <<= 1) {
        sm += __shfl_xor(sm, m, 16);
        sq += __shfl_xor(sq, m, 16);
      }
      const float mean = sm * (1.0f / 128.0f);
      const float var  = sq * (1.0f / 128.0f) - mean * mean;
      const float rstd = rsqrtf(var + LN_EPS);
      #pragma unroll
      for (int nt = 0; nt < 8; ++nt) {
        const int n = nt * 16 + lrow;
        out[(size_t)pm * HID + n] = (yv[nt] - mean) * rstd * gamma[n] + beta[n];
      }
    }
  }
}

// ---------------------------------------------------------------------------
extern "C" void kernel_launch(void* const* d_in, const int* in_sizes, int n_in,
                              void* d_out, int out_size, void* d_ws, size_t ws_size,
                              hipStream_t stream) {
  const float* xp    = (const float*)d_in[0];
  const float* xc    = (const float*)d_in[1];
  const int*   src   = (const int*)d_in[2];
  const int*   dst   = (const int*)d_in[3];
  const float* W1    = (const float*)d_in[4];
  const float* b1    = (const float*)d_in[5];
  const float* W2    = (const float*)d_in[6];
  const float* b2    = (const float*)d_in[7];
  const float* Wu    = (const float*)d_in[8];
  const float* bu    = (const float*)d_in[9];
  const float* gamma = (const float*)d_in[10];
  const float* beta  = (const float*)d_in[11];
  float* out  = (float*)d_out;

  float* ssum = (float*)d_ws;                          // [NPARENT*HID] f32
  float* cnt  = ssum + (size_t)NPARENT * HID;          // [NPARENT] f32

  const long nZero = (long)NPARENT * HID + NPARENT;
  zero_ws_kernel<<<512, 256, 0, stream>>>(ssum, nZero);
  count_kernel<<<(NEDGES + 255) / 256, 256, 0, stream>>>(src, cnt);

  const size_t edgeLds   = (size_t)(2 * 16384 + 8 * 2048) * 2;   // 96 KB
  const size_t parentLds = (size_t)16384 * 2;                    // 32 KB
  edge_kernel<<<512, 256, edgeLds, stream>>>(xp, xc, src, dst, W1, b1, W2, b2, ssum);
  parent_kernel<<<256, 256, parentLds, stream>>>(xp, ssum, cnt, Wu, bu, gamma, beta, out);
}
